// TensorDense_80367428043371
// MI455X (gfx1250) — matrix operand order repaired and checked by reading, not compiled, not run on hardware
//
#include <hip/hip_runtime.h>
#include <stdint.h>
#include <stddef.h>

typedef __attribute__((ext_vector_type(16))) _Float16 v16h;
typedef __attribute__((ext_vector_type(8)))  float    v8f;
typedef int v4i_ __attribute__((vector_size(16)));   // int4 for async-LDS builtin

#define B_DIM  2048
#define D_IN   4096
#define D_OUT  4096
#define KTILES (D_IN / 32)

// workspace layout (bytes):
//   Wt (f16) : D_OUT*D_IN*2       = 33,554,432   Wt[j*4096 + i] = W[i,j]
//   Xh (f16) : B_DIM*D_IN*2       = 16,777,216
//   T  (f32) : 16*16*16*16*32*4   =  8,388,608   T[x,a,y,b,q]
#define WT_BYTES ((size_t)D_OUT * D_IN * 2)
#define XH_BYTES ((size_t)B_DIM * D_IN * 2)
#define T_ELEMS  (16*16*16*16*32)

// gfx1250 async global->LDS copy path (ASYNCcnt), with compile-safe fallback.
#if defined(__has_builtin)
#if __has_builtin(__builtin_amdgcn_global_load_async_to_lds_b128) && \
    __has_builtin(__builtin_amdgcn_s_wait_asynccnt)
#define USE_ASYNC_LDS 1
#endif
#endif

#define AS1P(p) ((__attribute__((address_space(1))) v4i_*)(p))
#define AS3P(p) ((__attribute__((address_space(3))) v4i_*)(p))

// ---------------------------------------------------------------- x -> f16
__global__ void cvt_x_f16(const float* __restrict__ x, _Float16* __restrict__ xh, int n) {
  int t = blockIdx.x * blockDim.x + threadIdx.x;
  if (t < n) xh[t] = (_Float16)x[t];
}

// ------------------------------------------- T[x,a,y,b,q] = sum_p c1 * c2
__global__ void tt_merge12(const float* __restrict__ c1, const float* __restrict__ c2,
                           float* __restrict__ T) {
  int t = blockIdx.x * blockDim.x + threadIdx.x;   // 2^21 threads
  int q =  t        & 31;
  int b = (t >> 5)  & 15;
  int y = (t >> 9)  & 15;
  int a = (t >> 13) & 15;
  int x = (t >> 17) & 15;
  const float* c1p = c1 + ((x * 16 + a) << 5);          // c1[x,a,p], p stride 1
  const float* c2p = c2 + ((y * 16 + b) << 10) + q;     // c2[y,b,p,q], p stride 32
  float acc = 0.f;
#pragma unroll
  for (int p = 0; p < 32; ++p)
    acc = fmaf(c1p[p], c2p[p << 5], acc);
  T[t] = acc;
}

// -------------------------- Wt[j,i] = sum_q T[x,a,y,b,q] * c3[z,c,q] (f16)
__global__ void tt_merge3(const float* __restrict__ T, const float* __restrict__ c3,
                          _Float16* __restrict__ Wt) {
  int t = blockIdx.x * blockDim.x + threadIdx.x;   // 2^24 threads, t = j*4096 + i
  int i = t & 4095;
  int j = t >> 12;
  int x = i >> 8, y = (i >> 4) & 15, z = i & 15;
  int a = j >> 8, b = (j >> 4) & 15, c = j & 15;
  const float* Tp  = T  + (((((x * 16 + a) * 16 + y) * 16 + b)) << 5);
  const float* c3p = c3 + ((z * 16 + c) << 5);
  float acc = 0.f;
#pragma unroll
  for (int q = 0; q < 32; ++q)
    acc = fmaf(Tp[q], c3p[q], acc);
  Wt[t] = (_Float16)acc;
}

// ------------------------------------------------------------- WMMA GEMM
// out[2048,4096] = relu(Xh[2048,4096] @ W + bias), W given as Wt[j][i].
// Block: 256 thr / 8 waves, tile 128(M)x128(N), BK=32, double-buffered LDS.
// Wave grid 2x4 -> each wave owns 64x32 (4x2 fragments, 8 f32 accumulators).
union FragH { v16h v; uint4 q[2]; };

#define LDS_STRIDE 40   // halves; 80B row stride: 16B-aligned chunks, conflict-free

__global__ __launch_bounds__(256) void tt_gemm(
    const _Float16* __restrict__ Xh, const _Float16* __restrict__ Wt,
    const float* __restrict__ bias, float* __restrict__ out) {
  __shared__ __align__(16) _Float16 As[2][128 * LDS_STRIDE];
  __shared__ __align__(16) _Float16 Bs[2][128 * LDS_STRIDE];

  const int m0   = blockIdx.y * 128;
  const int n0   = blockIdx.x * 128;
  const int lane = threadIdx.x & 31;
  const int wid  = threadIdx.x >> 5;
  const int wm   = (wid >> 2) * 64;   // wave M offset: 0 / 64
  const int wn   = (wid & 3) * 32;    // wave N offset: 0..96
  const int lmod  = lane & 15;
  const int lhalf = lane >> 4;        // 0: lanes 0-15, 1: lanes 16-31

  // cooperative tile copy: 128 rows x 32 halves per tile, 16B per lane-chunk
  auto load_tiles = [&](int kt, int buf) {
    const int k0 = kt * 32;
#pragma unroll
    for (int it = 0; it < 2; ++it) {
      int idx = (int)threadIdx.x + it * 256;     // 0..511
      int row = idx >> 2;
      int seg = (idx & 3) << 3;                  // half offset 0,8,16,24
      const _Float16* ga = &Xh[(size_t)(m0 + row) * D_IN + k0 + seg];
      const _Float16* gb = &Wt[(size_t)(n0 + row) * D_IN + k0 + seg];
      _Float16* la = &As[buf][row * LDS_STRIDE + seg];
      _Float16* lb = &Bs[buf][row * LDS_STRIDE + seg];
#if defined(USE_ASYNC_LDS)
      __builtin_amdgcn_global_load_async_to_lds_b128(AS1P(ga), AS3P(la), 0, 0);
      __builtin_amdgcn_global_load_async_to_lds_b128(AS1P(gb), AS3P(lb), 0, 0);
#else
      *(uint4*)la = *(const uint4*)ga;
      *(uint4*)lb = *(const uint4*)gb;
#endif
      if (kt + 1 < KTILES) {                     // warm L2 for the tile after
        __builtin_prefetch(ga + 32, 0, 1);
        __builtin_prefetch(gb + 32, 0, 1);
      }
    }
  };

  v8f acc[4][2];
#pragma unroll
  for (int fm = 0; fm < 4; ++fm)
#pragma unroll
    for (int fn = 0; fn < 2; ++fn)
      acc[fm][fn] = (v8f){0.f, 0.f, 0.f, 0.f, 0.f, 0.f, 0.f, 0.f};

  load_tiles(0, 0);
#if defined(USE_ASYNC_LDS)
  __builtin_amdgcn_s_wait_asynccnt(0);
#endif
  __syncthreads();

  for (int kt = 0; kt < KTILES; ++kt) {
    const int cur = kt & 1;
    if (kt + 1 < KTILES) load_tiles(kt + 1, cur ^ 1);  // overlap DMA with math

    // A frag (16x32 f16): lane -> M=lmod; chunks at k-halves {s, s+16}, s=lhalf*8
    FragH af[4];
#pragma unroll
    for (int fm = 0; fm < 4; ++fm) {
      const _Float16* p = &As[cur][(wm + fm * 16 + lmod) * LDS_STRIDE + lhalf * 8];
      af[fm].q[0] = *(const uint4*)p;
      af[fm].q[1] = *(const uint4*)(p + 16);
    }
    // B frag (32x16 f16): lane -> N=lmod; chunks at k-halves {s, s+16}, s=lhalf*8
    FragH bf[2];
#pragma unroll
    for (int fn = 0; fn < 2; ++fn) {
      const _Float16* p = &Bs[cur][(wn + fn * 16 + lmod) * LDS_STRIDE + lhalf * 8];
      bf[fn].q[0] = *(const uint4*)p;
      bf[fn].q[1] = *(const uint4*)(p + 16);
    }

#pragma unroll
    for (int fm = 0; fm < 4; ++fm)
#pragma unroll
      for (int fn = 0; fn < 2; ++fn)
        acc[fm][fn] = __builtin_amdgcn_wmma_f32_16x16x32_f16(
            false, af[fm].v, false, bf[fn].v, (short)0, acc[fm][fn], false, false);

    if (kt + 1 < KTILES) {
#if defined(USE_ASYNC_LDS)
      __builtin_amdgcn_s_wait_asynccnt(0);       // my copies into buf^1 done
#endif
      __syncthreads();                           // everyone's copies + reads done
    }
  }

  // epilogue: C/D layout: lane -> N=lmod, reg r -> M=r (+8 for lanes 16-31)
#pragma unroll
  for (int fm = 0; fm < 4; ++fm) {
    const int row = m0 + wm + fm * 16 + lhalf * 8;
#pragma unroll
    for (int fn = 0; fn < 2; ++fn) {
      const int col = n0 + wn + fn * 16 + lmod;
      const float bv = bias[col];
#pragma unroll
      for (int r = 0; r < 8; ++r) {
        float v = acc[fm][fn][r] + bv;
        out[(size_t)(row + r) * D_OUT + col] = v > 0.f ? v : 0.f;
      }
    }
  }
}

extern "C" void kernel_launch(void* const* d_in, const int* in_sizes, int n_in,
                              void* d_out, int out_size, void* d_ws, size_t ws_size,
                              hipStream_t stream) {
  (void)in_sizes; (void)n_in; (void)out_size; (void)ws_size;
  const float* x    = (const float*)d_in[0];
  const float* c1   = (const float*)d_in[1];
  const float* c2   = (const float*)d_in[2];
  const float* c3   = (const float*)d_in[3];
  const float* bias = (const float*)d_in[4];
  float* out = (float*)d_out;

  char* ws = (char*)d_ws;
  _Float16* Wt = (_Float16*)ws;                         // 33.5 MB
  _Float16* Xh = (_Float16*)(ws + WT_BYTES);            // 16.8 MB
  float*    T  = (float*)(ws + WT_BYTES + XH_BYTES);    //  8.4 MB

  cvt_x_f16 <<<(B_DIM * D_IN) / 256, 256, 0, stream>>>(x, Xh, B_DIM * D_IN);
  tt_merge12<<<T_ELEMS / 256,        256, 0, stream>>>(c1, c2, T);
  tt_merge3 <<<(D_IN * D_OUT) / 256, 256, 0, stream>>>(T, c3, Wt);

  dim3 grid(D_OUT / 128, B_DIM / 128);   // 32 x 16 blocks
  tt_gemm<<<grid, 256, 0, stream>>>(Xh, Wt, bias, out);
}
